// TopkRouter_73443940761662
// MI455X (gfx1250) — matrix operand order repaired and checked by reading, not compiled, not run on hardware
//
#include <hip/hip_runtime.h>
#include <hip/hip_bf16.h>

#define EMBED 2048
#define NEXP  64
#define TOPK  8
#define TOK_PER_WAVE   16
#define WAVES_PER_BLK  8
#define TOK_PER_BLK    (TOK_PER_WAVE * WAVES_PER_BLK)   // 128 tokens / block

typedef __attribute__((ext_vector_type(16))) __bf16 v16bf;
typedef __attribute__((ext_vector_type(8)))  float  v8f;

// ---------------------------------------------------------------------------
// Pre-pass: convert router weights fp32 -> bf16 once into d_ws (L2-resident).
// ---------------------------------------------------------------------------
__global__ void convert_w_bf16(const float* __restrict__ W,
                               unsigned short* __restrict__ wb_raw, int n) {
    int i = blockIdx.x * blockDim.x + threadIdx.x;
    if (i < n) {
        __bf16* wb = (__bf16*)wb_raw;
        wb[i] = (__bf16)W[i];
    }
}

// ---------------------------------------------------------------------------
// Main router kernel: bf16 WMMA GEMM (16 tok x 64 experts per wave) + fused
// split-lane top-8 selection and masked softmax.
// ---------------------------------------------------------------------------
__global__ __launch_bounds__(256)
void router_kernel(const float* __restrict__ x,
                   const unsigned short* __restrict__ wb_raw,
                   const float* __restrict__ bias,
                   float* __restrict__ probs,
                   int* __restrict__ idx_out,
                   int ntok) {
    const __bf16* Wb = (const __bf16*)wb_raw;

    __shared__ float lds_logits[WAVES_PER_BLK][TOK_PER_WAVE][NEXP];   // 32 KB
    __shared__ int   lds_idx[WAVES_PER_BLK][TOK_PER_WAVE][TOPK];      //  4 KB
    __shared__ float cand_v[WAVES_PER_BLK][TOK_PER_WAVE][2 * TOPK];   //  8 KB
    __shared__ int   cand_i[WAVES_PER_BLK][TOK_PER_WAVE][2 * TOPK];   //  8 KB

    const int tid  = threadIdx.x;
    const int w    = tid >> 5;        // wave id in block (wave32)
    const int lane = tid & 31;
    const int row  = lane & 15;       // token-in-wave (A) / expert-in-frag (B,C)
    const int g    = lane >> 4;       // K-half selector (contiguous k = 16*g + s, same for A and B)

    const int m0 = (blockIdx.x * WAVES_PER_BLK + w) * TOK_PER_WAVE;
    if (blockIdx.x * TOK_PER_BLK >= ntok) return;   // uniform per block

    // A: this lane streams token row m0+row. Contiguous K-half 16*g (matches B).
    const float* ap = x + (size_t)(m0 + row) * EMBED;
    const int khalfA = g * 16;
    // B: this lane holds expert column 16*f + row, K-half 16*g (32x16 layout).
    const __bf16* bp0 = Wb + (size_t)row * EMBED + g * 16;

    v8f acc[4];
    #pragma unroll
    for (int f = 0; f < 4; ++f)
        #pragma unroll
        for (int r = 0; r < 8; ++r) acc[f][r] = 0.0f;

    #pragma unroll 2
    for (int k0 = 0; k0 < EMBED; k0 += 32) {
        const float* pa = ap + k0 + khalfA;
        float4 c0 = *(const float4*)(pa);
        float4 c1 = *(const float4*)(pa + 4);
        float4 c2 = *(const float4*)(pa + 8);
        float4 c3 = *(const float4*)(pa + 12);
        __builtin_prefetch(pa + 256, 0, 3);   // global_prefetch_b8, near-cache

        v16bf a;
        a[0]  = (__bf16)c0.x; a[1]  = (__bf16)c0.y; a[2]  = (__bf16)c0.z; a[3]  = (__bf16)c0.w;
        a[4]  = (__bf16)c1.x; a[5]  = (__bf16)c1.y; a[6]  = (__bf16)c1.z; a[7]  = (__bf16)c1.w;
        a[8]  = (__bf16)c2.x; a[9]  = (__bf16)c2.y; a[10] = (__bf16)c2.z; a[11] = (__bf16)c2.w;
        a[12] = (__bf16)c3.x; a[13] = (__bf16)c3.y; a[14] = (__bf16)c3.z; a[15] = (__bf16)c3.w;

        #pragma unroll
        for (int f = 0; f < 4; ++f) {
            v16bf bf = *(const v16bf*)(bp0 + (size_t)f * 16 * EMBED + k0);
            acc[f] = __builtin_amdgcn_wmma_f32_16x16x32_bf16(
                false, a, false, bf, (short)0, acc[f], false, false);
        }
    }

    // Bias per expert column held by this lane, then scatter logits to LDS.
    float bn[4];
    #pragma unroll
    for (int f = 0; f < 4; ++f) bn[f] = bias[f * 16 + row];

    #pragma unroll
    for (int f = 0; f < 4; ++f)
        #pragma unroll
        for (int r = 0; r < 8; ++r)   // C/D layout: M = r + 8*g, N = 16*f + row
            lds_logits[w][r + 8 * g][f * 16 + row] = acc[f][r] + bn[f];

    __syncthreads();

    // ---- Phase 1: split top-8 scan. Lanes t and t+16 each scan 32 experts of
    // token t and publish a locally-sorted (descending) top-8 candidate list.
    {
        const int t = row;                 // token handled by this lane
        const int h = g;                   // expert half: 0 -> [0,32), 1 -> [32,64)
        float lv[32];
        #pragma unroll
        for (int i = 0; i < 8; ++i)
            ((float4*)lv)[i] = ((const float4*)&lds_logits[w][t][h * 32])[i];

        unsigned int selm = 0u;
        #pragma unroll
        for (int j = 0; j < TOPK; ++j) {
            float bm = -3.402823466e38f; int bi = 0;
            #pragma unroll
            for (int i = 0; i < 32; ++i) {
                bool fresh = ((selm >> i) & 1u) == 0u;
                if (fresh && lv[i] > bm) { bm = lv[i]; bi = i; }
            }
            selm |= 1u << bi;
            cand_v[w][t][h * TOPK + j] = bm;
            cand_i[w][t][h * TOPK + j] = h * 32 + bi;   // global expert index
        }
    }

    __syncthreads();

    // ---- Phase 2: lanes 0..15 merge the two sorted 8-lists of their token,
    // run the masked softmax, and stage probs/indices for coalesced output.
    if (lane < TOK_PER_WAVE) {
        const int t = lane;
        int i = 0, j = 0;
        float vals[TOPK]; int ids[TOPK];
        #pragma unroll
        for (int k = 0; k < TOPK; ++k) {
            float av = (i < TOPK) ? cand_v[w][t][i]        : -3.402823466e38f;
            float bv = (j < TOPK) ? cand_v[w][t][TOPK + j] : -3.402823466e38f;
            if (av >= bv) {                    // tie -> low half -> lower index
                vals[k] = av; ids[k] = cand_i[w][t][i]; ++i;
            } else {
                vals[k] = bv; ids[k] = cand_i[w][t][TOPK + j]; ++j;
            }
        }

        const float mx = vals[0];              // descending -> max first
        float ex[TOPK]; float s = 0.0f;
        #pragma unroll
        for (int k = 0; k < TOPK; ++k) { ex[k] = __expf(vals[k] - mx); s += ex[k]; }
        const float inv = 1.0f / s;

        // probs row: zeros everywhere, softmax weights on the selected 8.
        #pragma unroll
        for (int q = 0; q < NEXP / 4; ++q)
            ((float4*)&lds_logits[w][t][0])[q] = make_float4(0.f, 0.f, 0.f, 0.f);
        #pragma unroll
        for (int k = 0; k < TOPK; ++k) lds_logits[w][t][ids[k]] = ex[k] * inv;
        #pragma unroll
        for (int k = 0; k < TOPK; ++k) lds_idx[w][t][k] = ids[k];
    }

    __syncthreads();

    // Coalesced writeback: 128 tok x 64 probs = 2048 float4; 1024 idx = 256 int4.
    float4* op = (float4*)probs + (size_t)blockIdx.x * (TOK_PER_BLK * NEXP / 4);
    const float4* lsrc = (const float4*)&lds_logits[0][0][0];
    #pragma unroll
    for (int q = 0; q < 8; ++q) op[tid + q * 256] = lsrc[tid + q * 256];

    int4* oi = (int4*)idx_out + (size_t)blockIdx.x * (TOK_PER_BLK * TOPK / 4);
    oi[tid] = ((const int4*)&lds_idx[0][0][0])[tid];
}

// ---------------------------------------------------------------------------
extern "C" void kernel_launch(void* const* d_in, const int* in_sizes, int n_in,
                              void* d_out, int out_size, void* d_ws, size_t ws_size,
                              hipStream_t stream) {
    const float* x = (const float*)d_in[0];   // [8,4096,2048] f32
    const float* W = (const float*)d_in[1];   // [64,2048]     f32
    const float* b = (const float*)d_in[2];   // [64]          f32

    const int ntok = in_sizes[0] / EMBED;     // 32768
    const int wlen = in_sizes[1];             // 131072

    float* probs = (float*)d_out;                          // [ntok,64] f32
    int*   idxo  = (int*)(probs + (size_t)ntok * NEXP);    // [ntok,8] i32 (raw bits)
    unsigned short* wb = (unsigned short*)d_ws;            // bf16 W scratch

    convert_w_bf16<<<(wlen + 255) / 256, 256, 0, stream>>>(W, wb, wlen);
    router_kernel<<<ntok / TOK_PER_BLK, 256, 0, stream>>>(x, wb, b, probs, idxo, ntok);
}
